// RGCN_78116865180365
// MI455X (gfx1250) — compile-verified, matrix operand order checked by reading, not run on hardware
//
#include <hip/hip_runtime.h>
#include <hip/hip_bf16.h>

#define N_NODES 100000
#define N_EDGES 1200000
#define N_REL   6

typedef __attribute__((ext_vector_type(16))) _Float16 v16h;
typedef __attribute__((ext_vector_type(8)))  _Float16 v8h;
typedef __attribute__((ext_vector_type(8)))  float    v8f;

// ---------------------------------------------------------------------------
// Kernel 1: scene MLP (22->32->32->64) fused with building X0 f16 [N,128]
//   cols 0..56  = pn_feat, cols 57..120 = scene embed, cols 121..127 = 0 pad
// ---------------------------------------------------------------------------
__global__ void k_mlp_build_x(const float* __restrict__ pn,
                              const float* __restrict__ pos,
                              const float* __restrict__ rot,
                              const float* __restrict__ osz,
                              const float* __restrict__ attr,
                              const float* __restrict__ sw1, const float* __restrict__ sb1,
                              const float* __restrict__ sw2, const float* __restrict__ sb2,
                              const float* __restrict__ sw3, const float* __restrict__ sb3,
                              _Float16* __restrict__ X)
{
    __shared__ float W1[22 * 32];
    __shared__ float W2[32 * 32];
    __shared__ float W3[32 * 64];
    __shared__ float B1[32], B2[32], B3[64];
    for (int t = threadIdx.x; t < 22 * 32; t += blockDim.x) W1[t] = sw1[t];
    for (int t = threadIdx.x; t < 32 * 32; t += blockDim.x) W2[t] = sw2[t];
    for (int t = threadIdx.x; t < 32 * 64; t += blockDim.x) W3[t] = sw3[t];
    if (threadIdx.x < 32) { B1[threadIdx.x] = sb1[threadIdx.x]; B2[threadIdx.x] = sb2[threadIdx.x]; }
    if (threadIdx.x < 64) B3[threadIdx.x] = sb3[threadIdx.x];
    __syncthreads();

    int i = blockIdx.x * blockDim.x + threadIdx.x;
    if (i >= N_NODES) return;

    float s[22];
    #pragma unroll
    for (int k = 0; k < 3; ++k)  s[k]      = pos[(size_t)i * 3 + k];
    #pragma unroll
    for (int k = 0; k < 4; ++k)  s[3 + k]  = rot[(size_t)i * 4 + k];
    #pragma unroll
    for (int k = 0; k < 3; ++k)  s[7 + k]  = osz[(size_t)i * 3 + k];
    #pragma unroll
    for (int k = 0; k < 12; ++k) s[10 + k] = attr[(size_t)i * 12 + k];

    float h1[32];
    #pragma unroll
    for (int j = 0; j < 32; ++j) h1[j] = B1[j];
    #pragma unroll
    for (int k = 0; k < 22; ++k) {
        float sv = s[k];
        #pragma unroll
        for (int j = 0; j < 32; ++j) h1[j] = fmaf(sv, W1[k * 32 + j], h1[j]);
    }
    #pragma unroll
    for (int j = 0; j < 32; ++j) h1[j] = fmaxf(h1[j], 0.f);

    float h2[32];
    #pragma unroll
    for (int j = 0; j < 32; ++j) h2[j] = B2[j];
    #pragma unroll
    for (int k = 0; k < 32; ++k) {
        float sv = h1[k];
        #pragma unroll
        for (int j = 0; j < 32; ++j) h2[j] = fmaf(sv, W2[k * 32 + j], h2[j]);
    }
    #pragma unroll
    for (int j = 0; j < 32; ++j) h2[j] = fmaxf(h2[j], 0.f);

    _Float16* xr = X + (size_t)i * 128;
    #pragma unroll
    for (int c = 0; c < 57; ++c) xr[c] = (_Float16)pn[(size_t)i * 57 + c];
    #pragma unroll
    for (int j = 0; j < 64; ++j) {
        float e = B3[j];
        #pragma unroll
        for (int k = 0; k < 32; ++k) e = fmaf(h2[k], W3[k * 64 + j], e);
        xr[57 + j] = (_Float16)e;
    }
    #pragma unroll
    for (int c = 121; c < 128; ++c) xr[c] = (_Float16)0.f;
}

// ---------------------------------------------------------------------------
// Edge-count kernels: cnt[node][rel] -> inv = 1/max(cnt,1), computed once
// ---------------------------------------------------------------------------
__global__ void k_zero_u32(unsigned* __restrict__ p, int n)
{
    int i = blockIdx.x * blockDim.x + threadIdx.x;
    if (i < n) p[i] = 0u;
}

__global__ void k_count(const int* __restrict__ ei, const int* __restrict__ et,
                        unsigned* __restrict__ cnt)
{
    int e = blockIdx.x * blockDim.x + threadIdx.x;
    if (e >= N_EDGES) return;
    int dst = ei[N_EDGES + e];
    int r   = et[e];
    atomicAdd(&cnt[dst * N_REL + r], 1u);
}

__global__ void k_invert(unsigned* __restrict__ c, float* __restrict__ inv, int n)
{
    int i = blockIdx.x * blockDim.x + threadIdx.x;
    if (i >= n) return;
    unsigned v = c[i];
    inv[i] = 1.0f / (float)(v > 0u ? v : 1u);   // in-place reinterpret is safe per-element
}

// ---------------------------------------------------------------------------
// Pack [rel_w(6,din,64) ; root_w(din,64)] f32 -> f16 B fragments pre-swizzled
// into the CDNA5 16-bit B-matrix (32x16) VGPR layout so that each wave lane
// loads one contiguous 32B v16h per fragment.
// Fragment (kt,ct): halves laid out [(kt*28+ct)*512 + lane*16 + i]
//   element i of lane L -> K = kt*32 + (L>=16?8:0) + (i&7) + (i>=8?16:0)
//                          N = ct*16 + (L&15)   (cols 0..383 = rel, 384..447 = root)
// ---------------------------------------------------------------------------
__global__ void k_pack_w(const float* __restrict__ relw, const float* __restrict__ rootw,
                         _Float16* __restrict__ Bsw, int din, int ktiles)
{
    int idx   = blockIdx.x * blockDim.x + threadIdx.x;
    int total = ktiles * 28 * 512;
    if (idx >= total) return;
    int i    = idx & 15;
    int lane = (idx >> 4) & 31;
    int frag = idx >> 9;
    int ct   = frag % 28;
    int kt   = frag / 28;
    int klocal = ((lane >> 4) << 3) + (i & 7) + ((i & 8) ? 16 : 0);
    int k   = kt * 32 + klocal;
    int col = ct * 16 + (lane & 15);
    float v = 0.f;
    if (k < din) {
        if (col < 384) v = relw[((size_t)(col >> 6) * din + k) * 64 + (col & 63)];
        else           v = rootw[(size_t)k * 64 + (col - 384)];
    }
    Bsw[idx] = (_Float16)v;
}

// ---------------------------------------------------------------------------
// WMMA GEMM: H[N,448] (f16) = X[N,kdim] (f16) @ Wcat[kdim,448]
// Block = 224 threads = 7 waves; wave w produces a 16x64 tile (4 accumulators).
// A fragment per CDNA5 16-bit A layout: lane L supplies row L%16,
//   halves [kk + (L/16)*8 .. +7] and [kk + 16 + (L/16)*8 .. +7].
// ---------------------------------------------------------------------------
__global__ void __launch_bounds__(224)
k_gemm448(const _Float16* __restrict__ X, const _Float16* __restrict__ Bsw,
          _Float16* __restrict__ H, int kdim)
{
    const int lane    = threadIdx.x & 31;
    const int wave    = threadIdx.x >> 5;      // 0..6 -> 64-col group
    const int rowTile = blockIdx.x;            // 16 rows per tile (N % 16 == 0)
    const int m       = lane & 15;
    const int hh      = lane >> 4;             // 0 or 1
    const _Float16* xrow = X + (size_t)(rowTile * 16 + m) * kdim;
    const int ktiles = kdim >> 5;

    v8f acc0 = {}, acc1 = {}, acc2 = {}, acc3 = {};
    for (int kt = 0; kt < ktiles; ++kt) {
        union { v16h v; v8h h[2]; } A;
        const int kk = kt * 32 + hh * 8;
        A.h[0] = *(const v8h*)(xrow + kk);
        A.h[1] = *(const v8h*)(xrow + kk + 16);
        const _Float16* bp = Bsw + ((size_t)kt * 28 + wave * 4) * 512 + lane * 16;
        v16h b0 = *(const v16h*)(bp);
        v16h b1 = *(const v16h*)(bp + 512);
        v16h b2 = *(const v16h*)(bp + 1024);
        v16h b3 = *(const v16h*)(bp + 1536);
        acc0 = __builtin_amdgcn_wmma_f32_16x16x32_f16(false, A.v, false, b0, (short)0, acc0, false, false);
        acc1 = __builtin_amdgcn_wmma_f32_16x16x32_f16(false, A.v, false, b1, (short)0, acc1, false, false);
        acc2 = __builtin_amdgcn_wmma_f32_16x16x32_f16(false, A.v, false, b2, (short)0, acc2, false, false);
        acc3 = __builtin_amdgcn_wmma_f32_16x16x32_f16(false, A.v, false, b3, (short)0, acc3, false, false);
    }

    // D layout: VGPR j -> row rowTile*16 + j + 8*(lane>=16), col = group + 16*ct + lane%16
    _Float16* hbase = H + (size_t)(rowTile * 16 + hh * 8) * 448 + wave * 64 + m;
    #pragma unroll
    for (int j = 0; j < 8; ++j) {
        hbase[(size_t)j * 448 +  0] = (_Float16)acc0[j];
        hbase[(size_t)j * 448 + 16] = (_Float16)acc1[j];
        hbase[(size_t)j * 448 + 32] = (_Float16)acc2[j];
        hbase[(size_t)j * 448 + 48] = (_Float16)acc3[j];
    }
}

// ---------------------------------------------------------------------------
// out[i][c] = root-transform (H cols 384..447) + bias
// ---------------------------------------------------------------------------
__global__ void k_init_out(const _Float16* __restrict__ H, const float* __restrict__ bias,
                           float* __restrict__ out)
{
    int idx = blockIdx.x * blockDim.x + threadIdx.x;
    if (idx >= N_NODES * 64) return;
    int c = idx & 63;
    out[idx] = (float)H[(size_t)(idx >> 6) * 448 + 384 + c] + bias[c];
}

// ---------------------------------------------------------------------------
// Edge scatter: 16 threads/edge, 4 cols/thread.
// out[dst][c] += H[src][r*64+c] * inv[dst][r]   (segment-mean)
// ---------------------------------------------------------------------------
__global__ void k_scatter(const int* __restrict__ ei, const int* __restrict__ et,
                          const _Float16* __restrict__ H, const float* __restrict__ inv,
                          float* __restrict__ out)
{
    int t = blockIdx.x * blockDim.x + threadIdx.x;
    int e = t >> 4;
    if (e >= N_EDGES) return;
    int sub = t & 15;
    int src = ei[e];
    int dst = ei[N_EDGES + e];
    int r   = et[e];
    float w = inv[dst * N_REL + r];
    const _Float16* hp = H + (size_t)src * 448 + r * 64 + sub * 4;
    float* op = out + (size_t)dst * 64 + sub * 4;
    #pragma unroll
    for (int k = 0; k < 4; ++k)
        atomicAdd(op + k, (float)hp[k] * w);
}

// ---------------------------------------------------------------------------
// ReLU + f16 convert for next layer's X (stride 64, layers 2/3 have din=64)
// ---------------------------------------------------------------------------
__global__ void k_relu_x(const float* __restrict__ out, _Float16* __restrict__ X)
{
    int idx = blockIdx.x * blockDim.x + threadIdx.x;
    if (idx >= N_NODES * 64) return;
    X[idx] = (_Float16)fmaxf(out[idx], 0.f);
}

// ---------------------------------------------------------------------------
extern "C" void kernel_launch(void* const* d_in, const int* in_sizes, int n_in,
                              void* d_out, int out_size, void* d_ws, size_t ws_size,
                              hipStream_t stream)
{
    const float* pn   = (const float*)d_in[0];
    const float* pos  = (const float*)d_in[1];
    const float* rot  = (const float*)d_in[2];
    const float* osz  = (const float*)d_in[3];
    const float* attr = (const float*)d_in[4];
    const int*   ei   = (const int*)d_in[5];
    const int*   et   = (const int*)d_in[6];
    const float* sw1 = (const float*)d_in[7];  const float* sb1 = (const float*)d_in[8];
    const float* sw2 = (const float*)d_in[9];  const float* sb2 = (const float*)d_in[10];
    const float* sw3 = (const float*)d_in[11]; const float* sb3 = (const float*)d_in[12];
    const float* rel_w[3]  = {(const float*)d_in[13], (const float*)d_in[16], (const float*)d_in[19]};
    const float* root_w[3] = {(const float*)d_in[14], (const float*)d_in[17], (const float*)d_in[20]};
    const float* bias[3]   = {(const float*)d_in[15], (const float*)d_in[18], (const float*)d_in[21]};

    // Workspace carve-up (~143 MB total)
    char* ws = (char*)d_ws;
    size_t off = 0;
    auto carve = [&](size_t bytes) -> void* {
        void* p = ws + off;
        off = (off + bytes + 255) & ~(size_t)255;
        return p;
    };
    _Float16* X    = (_Float16*)carve((size_t)N_NODES * 128 * 2);  // 25.6 MB
    _Float16* H    = (_Float16*)carve((size_t)N_NODES * 448 * 2);  // 89.6 MB
    float*    obuf = (float*)   carve((size_t)N_NODES * 64 * 4);   // 25.6 MB
    unsigned* cnt  = (unsigned*)carve((size_t)N_NODES * N_REL * 4);// 2.4 MB (reused as inv f32)
    _Float16* Wsw  = (_Float16*)carve((size_t)4 * 28 * 512 * 2);   // 114 KB
    (void)ws_size; (void)in_sizes; (void)n_in; (void)out_size;

    // Scene MLP + build X0 [N,128] f16
    k_mlp_build_x<<<(N_NODES + 255) / 256, 256, 0, stream>>>(
        pn, pos, rot, osz, attr, sw1, sb1, sw2, sb2, sw3, sb3, X);

    // Per-(node,rel) in-degree -> 1/max(cnt,1)
    k_zero_u32<<<(N_NODES * N_REL + 255) / 256, 256, 0, stream>>>(cnt, N_NODES * N_REL);
    k_count  <<<(N_EDGES + 255) / 256, 256, 0, stream>>>(ei, et, cnt);
    k_invert <<<(N_NODES * N_REL + 255) / 256, 256, 0, stream>>>(cnt, (float*)cnt, N_NODES * N_REL);

    const int din[3]  = {121, 64, 64};
    const int kdim[3] = {128, 64, 64};
    for (int l = 0; l < 3; ++l) {
        int ktiles = kdim[l] >> 5;
        int wtot   = ktiles * 28 * 512;
        k_pack_w<<<(wtot + 255) / 256, 256, 0, stream>>>(rel_w[l], root_w[l], Wsw, din[l], ktiles);
        k_gemm448<<<N_NODES / 16, 224, 0, stream>>>(X, Wsw, H, kdim[l]);

        float* outp = (l == 2) ? (float*)d_out : obuf;
        k_init_out<<<(N_NODES * 64 + 255) / 256, 256, 0, stream>>>(H, bias[l], outp);
        k_scatter <<<((size_t)N_EDGES * 16 + 255) / 256, 256, 0, stream>>>(
            ei, et, H, (const float*)cnt, outp);
        if (l < 2)
            k_relu_x<<<(N_NODES * 64 + 255) / 256, 256, 0, stream>>>(outp, X);
    }
}
